// MinGRUBlock_35459249996094
// MI455X (gfx1250) — matrix operand order repaired and checked by reading, not compiled, not run on hardware
//
#include <hip/hip_runtime.h>
#include <hip/hip_bf16.h>

typedef __attribute__((ext_vector_type(16))) _Float16 v16h;
typedef __attribute__((ext_vector_type(8)))  float    v8f;
typedef __attribute__((ext_vector_type(4)))  unsigned int u32x4;
typedef __attribute__((ext_vector_type(8)))  int i32x8;
typedef __attribute__((ext_vector_type(4)))  int i32x4;

#define BB 8
#define SS 4096
#define DD 1024
#define FF 3072
#define MTOT (BB*SS)          // 32768 rows
#define CH 32                 // scan chunks
#define CHLEN (SS/CH)         // 128 steps per chunk
#define LDA 40                // padded LDS stride in halves (64B + 16B pad)

#if __has_builtin(__builtin_amdgcn_tensor_load_to_lds) && __has_builtin(__builtin_amdgcn_s_wait_tensorcnt)
#define USE_TDM 1
#else
#define USE_TDM 0
#endif

struct alignas(16) H8 { _Float16 h[8]; };
union FragU { v16h v; unsigned int u[8]; };

__device__ __forceinline__ float sigf(float x) { return 1.0f / (1.0f + __expf(-x)); }
__device__ __forceinline__ float tanh_fast(float x) { return 2.0f * sigf(2.0f * x) - 1.0f; }

#if USE_TDM
// TDM 2-D tile load: tile_rows x 32 halves from row-major [nrows][ldk] f16 tensor
// into LDS, padded to LDA-half row stride. D# layout per CDNA5 ISA ch.8.
__device__ __forceinline__ void tdm_load_tile(const _Float16* g, unsigned lds_off,
                                              int ldk, int nrows, int tile_rows) {
    unsigned long long ga = (unsigned long long)(uintptr_t)g;
    u32x4 g0 = { 1u,                                   // count=1 (valid descriptor)
                 lds_off,                              // lds_addr
                 (unsigned)ga,                         // global_addr[31:0]
                 (unsigned)((ga >> 32) & 0x01FFFFFFu) | (2u << 30) }; // addr[56:32] | type=2
    i32x8 g1 = {
        (int)((1u << 16)            // data_size = 2 bytes
            | (1u << 20)            // pad_enable
            | (3u << 22)            // pad_interval: 16 DWORDs (one 32-half row)
            | (3u << 25)),          // pad_amount: 4 DWORDs (8 halves) -> LDA=40
        (int)(((unsigned)ldk & 0xFFFFu) << 16),                        // tensor_dim0 lo16
        (int)((((unsigned)ldk >> 16) & 0xFFFFu)
            | (((unsigned)nrows & 0xFFFFu) << 16)),                    // dim0 hi16 | dim1 lo16
        (int)((((unsigned)nrows >> 16) & 0xFFFFu) | (32u << 16)),      // dim1 hi16 | tile_dim0=32
        tile_rows,                                                     // tile_dim1, tile_dim2=0
        ldk,                                                           // tensor_dim0_stride lo32
        0, 0 };
    i32x4 z4 = {0, 0, 0, 0};
#if __clang_major__ >= 23
    i32x8 z8 = {0, 0, 0, 0, 0, 0, 0, 0};
    __builtin_amdgcn_tensor_load_to_lds(g0, g1, z4, z4, z8, 0);
#else
    __builtin_amdgcn_tensor_load_to_lds(g0, g1, z4, z4, 0);
#endif
}
#endif

// ---------------- f32 -> f16 transposed convert: dst[c*R + r] = src[r*C + c] ----------------
__global__ void cvt_tr_f16_kernel(_Float16* __restrict__ dst, const float* __restrict__ src,
                                  int R, int C) {
    int n = R * C;
    int stride = gridDim.x * blockDim.x;
    for (int i = blockIdx.x * blockDim.x + threadIdx.x; i < n; i += stride) {
        int r = i / C, c = i % C;
        dst[(size_t)c * R + r] = (_Float16)src[i];
    }
}

// ---------------- RMSNorm: one block per token row ----------------
__global__ void rmsnorm_f16_kernel(_Float16* __restrict__ dst, const float* __restrict__ src,
                                   const float* __restrict__ w) {
    __shared__ float red[256];
    int row = blockIdx.x;
    int tid = threadIdx.x;
    const float* sr = src + (size_t)row * DD;
    float s = 0.f;
    #pragma unroll
    for (int i = tid; i < DD; i += 256) { float v = sr[i]; s += v * v; }
    red[tid] = s; __syncthreads();
    for (int off = 128; off > 0; off >>= 1) {
        if (tid < off) red[tid] += red[tid + off];
        __syncthreads();
    }
    float rms = rsqrtf(red[0] * (1.0f / DD) + 1e-6f);
    _Float16* dr = dst + (size_t)row * DD;
    #pragma unroll
    for (int i = tid; i < DD; i += 256) dr[i] = (_Float16)(sr[i] * rms * w[i]);
}

// ---------------- fused WMMA GEMM, (64*MT)x64 block tile, 8 waves, double-buffered ----------
// Each wave computes an MTx2 grid of 16x16 C tiles; MT chosen per mode to stay spill-free:
//   MODE 0: NM=3, MT=1 (48 acc VGPRs)   -> x_scan (f32), a (f32), with biases
//   MODE 1: NM=2, MT=2 (64 acc VGPRs)   -> t = silu(gate)*up (f16)
//   MODE 2: NM=1, MT=2 (32 acc VGPRs)   -> out += acc (residual add, f32, in-place)
// All operands row-major-by-K: A is [M][K], B* are pre-transposed weights [N][K].
template <int MODE>
__global__ __launch_bounds__(256)
void gemm_wmma_kernel(const _Float16* __restrict__ A,
                      const _Float16* __restrict__ B0,
                      const _Float16* __restrict__ B1,
                      const _Float16* __restrict__ B2,
                      int K, int ldb,
                      float* __restrict__ out0, float* __restrict__ out1,
                      _Float16* __restrict__ out16,
                      const float* __restrict__ bg, const float* __restrict__ bv,
                      const float* __restrict__ bd) {
    constexpr int NM = (MODE == 0) ? 3 : (MODE == 1) ? 2 : 1;
    constexpr int MT = (MODE == 0) ? 1 : 2;          // m-tiles per wave
    constexpr int MROWS = 64 * MT;                   // block tile rows
    __shared__ _Float16 sA[2][MROWS * LDA];
    __shared__ _Float16 sB[2][NM][64 * LDA];

    const int tid = threadIdx.x;
    const int Nb = blockIdx.x * 64;
    const int Mb = blockIdx.y * MROWS;

    const int w    = tid >> 5;       // wave 0..7
    const int lane = tid & 31;
    const int l15  = lane & 15;
    const bool hi  = lane >= 16;
    const int m0 = (w & 3) * MT;     // first of MT m-tiles (16-row tiles)
    const int n0 = (w >> 2) * 2;     // first of 2 n-tiles

    const _Float16* baseA = A + (size_t)Mb * K;
    const _Float16* baseB[NM];
    baseB[0] = B0 + (size_t)Nb * K;
    if constexpr (NM >= 2) baseB[1] = B1 + (size_t)Nb * K;
    if constexpr (NM >= 3) baseB[2] = B2 + (size_t)Nb * K;

    v8f acc[NM][2][MT];              // [mat][nt][mt]
    #pragma unroll
    for (int m = 0; m < NM; ++m)
        #pragma unroll
        for (int i = 0; i < 2; ++i)
            #pragma unroll
            for (int j = 0; j < MT; ++j) acc[m][i][j] = (v8f)0.0f;

    const int srow = tid >> 2, scg = tid & 3;        // staging: rows x (4 x 8 halves)

    // ---- preamble: stage K-tile 0 into buffer 0 ----
    int buf = 0;
#if USE_TDM
    if (tid < 32) {
        tdm_load_tile(baseA, (unsigned)(uintptr_t)&sA[0][0], K, MTOT, MROWS);
        #pragma unroll
        for (int t = 0; t < NM; ++t)
            tdm_load_tile(baseB[t], (unsigned)(uintptr_t)&sB[0][t][0], K, ldb, 64);
        __builtin_amdgcn_s_wait_tensorcnt(0);
    }
#else
    {
        H8 ra[MT], rb[NM];
        #pragma unroll
        for (int rep = 0; rep < MT; ++rep)
            ra[rep] = *reinterpret_cast<const H8*>(baseA + (size_t)(srow + rep * 64) * K + scg * 8);
        #pragma unroll
        for (int t = 0; t < NM; ++t)
            rb[t] = *reinterpret_cast<const H8*>(baseB[t] + (size_t)srow * K + scg * 8);
        #pragma unroll
        for (int rep = 0; rep < MT; ++rep)
            *reinterpret_cast<H8*>(&sA[0][(srow + rep * 64) * LDA + scg * 8]) = ra[rep];
        #pragma unroll
        for (int t = 0; t < NM; ++t)
            *reinterpret_cast<H8*>(&sB[0][t][srow * LDA + scg * 8]) = rb[t];
    }
#endif
    __syncthreads();

    for (int k0 = 0; k0 < K; k0 += 32) {
        const int nxt = buf ^ 1;
        const bool hn = (k0 + 32) < K;
#if USE_TDM
        if (hn && tid < 32) {
            tdm_load_tile(baseA + k0 + 32, (unsigned)(uintptr_t)&sA[nxt][0], K, MTOT, MROWS);
            #pragma unroll
            for (int t = 0; t < NM; ++t)
                tdm_load_tile(baseB[t] + k0 + 32, (unsigned)(uintptr_t)&sB[nxt][t][0], K, ldb, 64);
        }
#else
        H8 ra[MT], rb[NM];
        if (hn) {
            #pragma unroll
            for (int rep = 0; rep < MT; ++rep)
                ra[rep] = *reinterpret_cast<const H8*>(
                    baseA + (size_t)(srow + rep * 64) * K + k0 + 32 + scg * 8);
            #pragma unroll
            for (int t = 0; t < NM; ++t)
                rb[t] = *reinterpret_cast<const H8*>(
                    baseB[t] + (size_t)srow * K + k0 + 32 + scg * 8);
        }
#endif
        // ---- load ALL fragments for this K-step up front (one big DS clause) ----
        // A fragment gather: lanes 0-15 -> K {0..7,16..23}; lanes 16-31 -> K {8..15,24..31}
        FragU fa[MT];
        const int kb1 = hi ? 4 : 0;          // uint units
        #pragma unroll
        for (int mt = 0; mt < MT; ++mt) {
            const unsigned int* pa = reinterpret_cast<const unsigned int*>(
                &sA[buf][((m0 + mt) * 16 + l15) * LDA]);
            #pragma unroll
            for (int i = 0; i < 4; ++i) {
                fa[mt].u[i]     = pa[kb1 + i];
                fa[mt].u[4 + i] = pa[kb1 + 8 + i];
            }
        }
        FragU fb[NM][2];
        const int kb = hi ? 4 : 0;
        #pragma unroll
        for (int mat = 0; mat < NM; ++mat)
            #pragma unroll
            for (int nt = 0; nt < 2; ++nt) {
                const unsigned int* pb = reinterpret_cast<const unsigned int*>(
                    &sB[buf][mat][((n0 + nt) * 16 + l15) * LDA]);
                #pragma unroll
                for (int i = 0; i < 8; ++i) fb[mat][nt].u[i] = pb[kb + i + (i & 4)];
            }
        // ---- WMMA chain ----
        #pragma unroll
        for (int mat = 0; mat < NM; ++mat)
            #pragma unroll
            for (int nt = 0; nt < 2; ++nt)
                #pragma unroll
                for (int mt = 0; mt < MT; ++mt)
                    acc[mat][nt][mt] = __builtin_amdgcn_wmma_f32_16x16x32_f16(
                        false, fa[mt].v, false, fb[mat][nt].v, (short)0,
                        acc[mat][nt][mt], false, false);
#if USE_TDM
        if (hn && tid < 32) __builtin_amdgcn_s_wait_tensorcnt(0);
#else
        if (hn) {
            #pragma unroll
            for (int rep = 0; rep < MT; ++rep)
                *reinterpret_cast<H8*>(&sA[nxt][(srow + rep * 64) * LDA + scg * 8]) = ra[rep];
            #pragma unroll
            for (int t = 0; t < NM; ++t)
                *reinterpret_cast<H8*>(&sB[nxt][t][srow * LDA + scg * 8]) = rb[t];
        }
#endif
        __syncthreads();
        buf = nxt;
    }

    // ---------------- epilogue ----------------
    #pragma unroll
    for (int nt = 0; nt < 2; ++nt) {
        const int n = Nb + (n0 + nt) * 16 + l15;
        float b0v = 0.f, b1v = 0.f, b2v = 0.f;
        if constexpr (MODE == 0) { b0v = bg[n]; b1v = bv[n]; b2v = bd[n]; }
        #pragma unroll
        for (int mt = 0; mt < MT; ++mt) {
            #pragma unroll
            for (int r = 0; r < 8; ++r) {
                const int m = Mb + (m0 + mt) * 16 + r + (hi ? 8 : 0);
                const size_t idx = (size_t)m * ldb + n;
                if constexpr (MODE == 0) {
                    float g = acc[0][nt][mt][r] + b0v;
                    float v = acc[1][nt][mt][r] + b1v;
                    float d = acc[2][nt][mt][r] + b2v;
                    out0[idx] = sigf(g) * tanh_fast(v);           // x_scan
                    out1[idx] = 0.001f + 0.998f * sigf(d);        // a
                } else if constexpr (MODE == 1) {
                    float g = acc[0][nt][mt][r];
                    float u = acc[1][nt][mt][r];
                    out16[idx] = (_Float16)(g * sigf(g) * u);     // silu(gate)*up
                } else {
                    out0[idx] = out0[idx] + acc[0][nt][mt][r];    // residual add
                }
            }
        }
    }
}

// ---------------- scan phase 1: per-chunk reduce (A_prod, H_partial) ----------------
__global__ void scan_phase1_kernel(const float* __restrict__ xs, const float* __restrict__ a,
                                   float* __restrict__ aggA, float* __restrict__ aggH) {
    const int DG = DD / 256;
    int dg = blockIdx.x % DG;
    int ch = (blockIdx.x / DG) % CH;
    int b  = blockIdx.x / (DG * CH);
    int d  = dg * 256 + threadIdx.x;
    float Ap = 1.f, Hp = 0.f;
    size_t base = ((size_t)b * SS + (size_t)ch * CHLEN) * DD + d;
    for (int t = 0; t < CHLEN; ++t) {
        float at = a[base], xt = xs[base];
        Hp = at * Hp + xt;
        Ap *= at;
        base += DD;
    }
    size_t chan = (size_t)b * DD + d;
    aggA[(size_t)ch * (BB * DD) + chan] = Ap;
    aggH[(size_t)ch * (BB * DD) + chan] = Hp;
}

// ---------------- scan phase 2: cross-chunk exclusive prefix ----------------
__global__ void scan_phase2_kernel(const float* __restrict__ aggA, const float* __restrict__ aggH,
                                   float* __restrict__ h0) {
    size_t chan = (size_t)blockIdx.x * 256 + threadIdx.x;   // B*D channels
    float runH = 0.f;
    for (int ch = 0; ch < CH; ++ch) {
        size_t idx = (size_t)ch * (BB * DD) + chan;
        h0[idx] = runH;
        runH = aggA[idx] * runH + aggH[idx];
    }
}

// ---------------- scan phase 3: apply + residual into d_out ----------------
__global__ void scan_phase3_kernel(const float* __restrict__ xs, const float* __restrict__ a,
                                   const float* __restrict__ h0, const float* __restrict__ x,
                                   float* __restrict__ out) {
    const int DG = DD / 256;
    int dg = blockIdx.x % DG;
    int ch = (blockIdx.x / DG) % CH;
    int b  = blockIdx.x / (DG * CH);
    int d  = dg * 256 + threadIdx.x;
    size_t chan = (size_t)b * DD + d;
    float h = h0[(size_t)ch * (BB * DD) + chan];
    size_t base = ((size_t)b * SS + (size_t)ch * CHLEN) * DD + d;
    for (int t = 0; t < CHLEN; ++t) {
        h = a[base] * h + xs[base];
        out[base] = x[base] + h;
        base += DD;
    }
}

extern "C" void kernel_launch(void* const* d_in, const int* in_sizes, int n_in,
                              void* d_out, int out_size, void* d_ws, size_t ws_size,
                              hipStream_t stream) {
    const float* x         = (const float*)d_in[0];
    const float* w_rms_mix = (const float*)d_in[1];
    const float* w_rms_ffn = (const float*)d_in[2];
    const float* Wg        = (const float*)d_in[3];
    const float* bg        = (const float*)d_in[4];
    const float* Wv        = (const float*)d_in[5];
    const float* bv        = (const float*)d_in[6];
    const float* Wd        = (const float*)d_in[7];
    const float* bd        = (const float*)d_in[8];
    const float* W_gate    = (const float*)d_in[9];
    const float* W_up      = (const float*)d_in[10];
    const float* W_out     = (const float*)d_in[11];
    float* out = (float*)d_out;

    char* ws = (char*)d_ws;
    const size_t MB = 1024ull * 1024ull;
    _Float16* XN16   = (_Float16*)(ws + 0);          //  64 MB (xn, later yn)
    float*    XSCAN  = (float*)(ws + 64 * MB);       // 128 MB
    float*    ABUF   = (float*)(ws + 192 * MB);      // 128 MB
    _Float16* T16    = (_Float16*)(ws + 64 * MB);    // 192 MB (aliases XSCAN+ABUF, used post-scan)
    float*    AGG_A  = (float*)(ws + 320 * MB);
    float*    AGG_H  = (float*)(ws + 321 * MB);
    float*    H0     = (float*)(ws + 322 * MB);
    _Float16* WG16   = (_Float16*)(ws + 323 * MB);   // transposed [1024][1024] f16
    _Float16* WV16   = (_Float16*)(ws + 325 * MB);
    _Float16* WD16   = (_Float16*)(ws + 327 * MB);
    _Float16* WGATE16= (_Float16*)(ws + 329 * MB);   // transposed [3072][1024] f16
    _Float16* WUP16  = (_Float16*)(ws + 335 * MB);
    _Float16* WOUT16 = (_Float16*)(ws + 341 * MB);   // transposed [1024][3072] f16

    // weight conversion + transpose (dst[N][K] from src[K][N])
    cvt_tr_f16_kernel<<<2048, 256, 0, stream>>>(WG16, Wg, DD, DD);
    cvt_tr_f16_kernel<<<2048, 256, 0, stream>>>(WV16, Wv, DD, DD);
    cvt_tr_f16_kernel<<<2048, 256, 0, stream>>>(WD16, Wd, DD, DD);
    cvt_tr_f16_kernel<<<4096, 256, 0, stream>>>(WGATE16, W_gate, DD, FF);
    cvt_tr_f16_kernel<<<4096, 256, 0, stream>>>(WUP16, W_up, DD, FF);
    cvt_tr_f16_kernel<<<4096, 256, 0, stream>>>(WOUT16, W_out, FF, DD);

    // xn = rmsnorm(x) -> f16
    rmsnorm_f16_kernel<<<MTOT, 256, 0, stream>>>(XN16, x, w_rms_mix);

    // mixer GEMM: g,v,d fused -> x_scan, a   (MT=1 -> 64-row blocks)
    {
        dim3 grid(DD / 64, MTOT / 64);
        gemm_wmma_kernel<0><<<grid, 256, 0, stream>>>(XN16, WG16, WV16, WD16, DD, DD,
                                                      XSCAN, ABUF, nullptr, bg, bv, bd);
    }

    // chunked linear scan + residual
    scan_phase1_kernel<<<BB * CH * (DD / 256), 256, 0, stream>>>(XSCAN, ABUF, AGG_A, AGG_H);
    scan_phase2_kernel<<<(BB * DD) / 256, 256, 0, stream>>>(AGG_A, AGG_H, H0);
    scan_phase3_kernel<<<BB * CH * (DD / 256), 256, 0, stream>>>(XSCAN, ABUF, H0, x, out);

    // yn = rmsnorm(x2) -> f16 (reuse XN16)
    rmsnorm_f16_kernel<<<MTOT, 256, 0, stream>>>(XN16, out, w_rms_ffn);

    // FFN GEMM 1: gate/up fused + SwiGLU -> t (f16)   (MT=2 -> 128-row blocks)
    {
        dim3 grid(FF / 64, MTOT / 128);
        gemm_wmma_kernel<1><<<grid, 256, 0, stream>>>(XN16, WGATE16, WUP16, nullptr, DD, FF,
                                                      nullptr, nullptr, T16, nullptr, nullptr, nullptr);
    }

    // FFN GEMM 2: t @ W_out, residual add into d_out   (MT=2 -> 128-row blocks)
    {
        dim3 grid(DD / 64, MTOT / 128);
        gemm_wmma_kernel<2><<<grid, 256, 0, stream>>>(T16, WOUT16, nullptr, nullptr, FF, DD,
                                                      out, nullptr, nullptr, nullptr, nullptr, nullptr);
    }
}
